// AFTAttention_29437705847475
// MI455X (gfx1250) — compile-verified, matrix operand order checked by reading, not run on hardware
//
#include <hip/hip_runtime.h>
#include <hip/hip_bf16.h>

// ---------------------------------------------------------------------------
// AFT attention for MI455X (gfx1250).
//  * bf16 WMMA GEMMs (v_wmma_f32_16x16x32_bf16), f32 accumulate.
//  * Both GEMM operands K-contiguous (B pre-transposed) -> pure b128 staging.
//  * Async global->LDS tile copies (ASYNCcnt) in the GEMM hot loop.
//  * Non-temporal hints on the write-once/read-once f32 streams.
// ---------------------------------------------------------------------------

#define USE_ASYNC_COPY 1

typedef __bf16 bf16_t;
typedef bf16_t v16bf __attribute__((ext_vector_type(16)));
typedef bf16_t v8bf  __attribute__((ext_vector_type(8)));
typedef float  v8f   __attribute__((ext_vector_type(8)));
typedef unsigned short u16x8 __attribute__((ext_vector_type(8)));

__device__ __forceinline__ unsigned short f32_to_bf16_bits(float f) {
  unsigned int u = __float_as_uint(f);
  u += 0x7FFFu + ((u >> 16) & 1u);   // round-to-nearest-even
  return (unsigned short)(u >> 16);
}

// ---------------------------------------------------------------------------
// C[f32, MxN] = A[bf16, MxK row-major] x BT[bf16, NxK row-major]^T
// Block tile 128x128, K-step 32; 8 waves x (64x32) = 4x2 WMMA accumulators.
// LDS double-buffered; padded stride 40 halves (80 B) keeps every fragment
// ds_load_b128 16B-aligned and bank-conflict free.
// triangular!=0: A lower-triangular in K (expbias) -> clip K at rowTile+BM.
// ---------------------------------------------------------------------------
#define BM 128
#define BN 128
#define BK 32
#define LDSS 40

__global__ __launch_bounds__(256)
void aft_gemm_bf16_wmma(const unsigned short* __restrict__ A,
                        const unsigned short* __restrict__ BT,
                        float* __restrict__ C,
                        int K, int lda, int ldb, int ldc,
                        long long strideA, long long strideB, long long strideC,
                        int triangular)
{
  __shared__ __align__(16) unsigned short shA[2][BM * LDSS];
  __shared__ __align__(16) unsigned short shB[2][BN * LDSS];

  const int tid = threadIdx.x;
  const long long bz = blockIdx.z;
  A  += bz * strideA;
  BT += bz * strideB;
  C  += bz * strideC;

  const int rowBase = blockIdx.y * BM;
  const int colBase = blockIdx.x * BN;
  int kEnd = K;
  if (triangular) { int ke = rowBase + BM; kEnd = (ke < K) ? ke : K; }
  const int nt = kEnd / BK;

  // 512 chunks of 8 halves per 128x32 tile; each thread moves chunk tid, tid+256
  const int c0 = tid, c1 = tid + 256;
  const int r0c = c0 >> 2, o0c = (c0 & 3) * 8;
  const int r1c = c1 >> 2, o1c = (c1 & 3) * 8;

#if USE_ASYNC_COPY
  auto issueTile = [&](int kt, int buf) {
    const unsigned short* ga0 = A  + (long long)(rowBase + r0c) * lda + kt * BK + o0c;
    const unsigned short* ga1 = A  + (long long)(rowBase + r1c) * lda + kt * BK + o1c;
    const unsigned short* gb0 = BT + (long long)(colBase + r0c) * ldb + kt * BK + o0c;
    const unsigned short* gb1 = BT + (long long)(colBase + r1c) * ldb + kt * BK + o1c;
    unsigned la0 = (unsigned)(uintptr_t)&shA[buf][r0c * LDSS + o0c];
    unsigned la1 = (unsigned)(uintptr_t)&shA[buf][r1c * LDSS + o1c];
    unsigned lb0 = (unsigned)(uintptr_t)&shB[buf][r0c * LDSS + o0c];
    unsigned lb1 = (unsigned)(uintptr_t)&shB[buf][r1c * LDSS + o1c];
    asm volatile(
        "global_load_async_to_lds_b128 %0, %4, off\n"
        "global_load_async_to_lds_b128 %1, %5, off\n"
        "global_load_async_to_lds_b128 %2, %6, off\n"
        "global_load_async_to_lds_b128 %3, %7, off\n"
        :
        : "v"(la0), "v"(la1), "v"(lb0), "v"(lb1),
          "v"((unsigned long long)(uintptr_t)ga0),
          "v"((unsigned long long)(uintptr_t)ga1),
          "v"((unsigned long long)(uintptr_t)gb0),
          "v"((unsigned long long)(uintptr_t)gb1)
        : "memory");
  };
#else
  u16x8 ra0, ra1, rb0, rb1;
  auto loadGlobal = [&](int kt) {
    ra0 = *(const u16x8*)(A  + (long long)(rowBase + r0c) * lda + kt * BK + o0c);
    ra1 = *(const u16x8*)(A  + (long long)(rowBase + r1c) * lda + kt * BK + o1c);
    rb0 = *(const u16x8*)(BT + (long long)(colBase + r0c) * ldb + kt * BK + o0c);
    rb1 = *(const u16x8*)(BT + (long long)(colBase + r1c) * ldb + kt * BK + o1c);
  };
  auto storeLds = [&](int buf) {
    *(u16x8*)&shA[buf][r0c * LDSS + o0c] = ra0;
    *(u16x8*)&shA[buf][r1c * LDSS + o1c] = ra1;
    *(u16x8*)&shB[buf][r0c * LDSS + o0c] = rb0;
    *(u16x8*)&shB[buf][r1c * LDSS + o1c] = rb1;
  };
#endif

  // wave tiling: 2x4 wave grid over the 128x128 block tile
  const int lane  = tid & 31;
  const int wave  = tid >> 5;
  const int wm    = (wave >> 2) * 64;
  const int wn    = (wave & 3) * 32;
  const int fr    = lane & 15;          // row (A) / col (B) within 16
  const int khalf = lane >> 4;          // 0: K 0-7 & 16-23, 1: K 8-15 & 24-31
  const int koff  = khalf * 8;

  v8f acc[4][2] = {};

#if USE_ASYNC_COPY
  issueTile(0, 0);
  asm volatile("s_wait_asynccnt 0" ::: "memory");
#else
  loadGlobal(0);
  storeLds(0);
#endif
  __syncthreads();

  for (int t = 0; t < nt; ++t) {
    const int cur = t & 1;
#if USE_ASYNC_COPY
    if (t + 1 < nt) issueTile(t + 1, cur ^ 1);   // async fills other buffer
#else
    if (t + 1 < nt) loadGlobal(t + 1);
#endif

    v16bf afrag[4], bfrag[2];
    #pragma unroll
    for (int mi = 0; mi < 4; ++mi) {
      const unsigned short* p = &shA[cur][(wm + mi * 16 + fr) * LDSS + koff];
      v8bf lo = __builtin_bit_cast(v8bf, *(const u16x8*)p);
      v8bf hi = __builtin_bit_cast(v8bf, *(const u16x8*)(p + 16));
      #pragma unroll
      for (int e = 0; e < 8; ++e) { afrag[mi][e] = lo[e]; afrag[mi][8 + e] = hi[e]; }
    }
    #pragma unroll
    for (int ni = 0; ni < 2; ++ni) {
      const unsigned short* p = &shB[cur][(wn + ni * 16 + fr) * LDSS + koff];
      v8bf lo = __builtin_bit_cast(v8bf, *(const u16x8*)p);
      v8bf hi = __builtin_bit_cast(v8bf, *(const u16x8*)(p + 16));
      #pragma unroll
      for (int e = 0; e < 8; ++e) { bfrag[ni][e] = lo[e]; bfrag[ni][8 + e] = hi[e]; }
    }

    #pragma unroll
    for (int mi = 0; mi < 4; ++mi)
      #pragma unroll
      for (int ni = 0; ni < 2; ++ni)
        acc[mi][ni] = __builtin_amdgcn_wmma_f32_16x16x32_bf16(
            false, afrag[mi], false, bfrag[ni], (short)0, acc[mi][ni], false, false);

    if (t + 1 < nt) {
#if USE_ASYNC_COPY
      asm volatile("s_wait_asynccnt 0" ::: "memory");
#else
      storeLds(cur ^ 1);
#endif
      __syncthreads();
    }
  }

  // epilogue: C/D layout VGPR r -> M = 8*khalf + r, N = lane&15 ; NT stores
  #pragma unroll
  for (int mi = 0; mi < 4; ++mi) {
    #pragma unroll
    for (int ni = 0; ni < 2; ++ni) {
      const int rr = rowBase + wm + mi * 16 + khalf * 8;
      const int cc = colBase + wn + ni * 16 + fr;
      #pragma unroll
      for (int r = 0; r < 8; ++r)
        __builtin_nontemporal_store(acc[mi][ni][r], &C[(long long)(rr + r) * ldc + cc]);
    }
  }
}

// ---------------------------------------------------------------------------
// f32 -> bf16 stream convert (read-once input: NT load)
// ---------------------------------------------------------------------------
__global__ __launch_bounds__(256)
void aft_cvt_bf16(const float* __restrict__ in, unsigned short* __restrict__ out, long long n)
{
  long long i = (long long)blockIdx.x * 256 + threadIdx.x;
  if (i < n) out[i] = f32_to_bf16_bits(__builtin_nontemporal_load(in + i));
}

// ---------------------------------------------------------------------------
// Transpose-convert: in[K][N] f32 -> out[N][K] bf16 (32x32 LDS tiles)
// ---------------------------------------------------------------------------
__global__ __launch_bounds__(256)
void aft_transpose_cvt(const float* __restrict__ in, unsigned short* __restrict__ out,
                       int K, int N)
{
  __shared__ unsigned short tile[32][33];
  const int tx = threadIdx.x, ty = threadIdx.y;
  const int n0 = blockIdx.x * 32, k0 = blockIdx.y * 32;
  #pragma unroll
  for (int i = 0; i < 4; ++i) {
    int k = k0 + ty + 8 * i;
    tile[ty + 8 * i][tx] =
        f32_to_bf16_bits(__builtin_nontemporal_load(&in[(long long)k * N + n0 + tx]));
  }
  __syncthreads();
  #pragma unroll
  for (int i = 0; i < 4; ++i) {
    int n = n0 + ty + 8 * i;
    out[(long long)n * K + k0 + tx] = tile[tx][ty + 8 * i];
  }
}

// ---------------------------------------------------------------------------
// Kmax[bt] = max_d k[bt][d]
// ---------------------------------------------------------------------------
__global__ __launch_bounds__(256)
void aft_rowmax_k(const float* __restrict__ proj, float* __restrict__ Kmax, int T, int D)
{
  const int t = blockIdx.x, b = blockIdx.y, tid = threadIdx.x;
  const long long bt = (long long)b * T + t;
  const float* krow = proj + bt * (3LL * D) + D;
  float m = -INFINITY;
  for (int d = tid; d < D; d += 256) m = fmaxf(m, krow[d]);
  __shared__ float red[256];
  red[tid] = m; __syncthreads();
  for (int s = 128; s > 0; s >>= 1) {
    if (tid < s) red[tid] = fmaxf(red[tid], red[tid + s]);
    __syncthreads();
  }
  if (tid == 0) Kmax[bt] = red[0];
}

// ---------------------------------------------------------------------------
// KVEt[b][d][t]   = bf16(exp(k-Kmax)*v)   (d in [0,D))
// KVEt[b][D+d][t] = bf16(exp(k-Kmax))     -> K-contiguous B operand for GEMM3
// Fused exp + 32x32 tile transpose; coalesced reads (along d) and writes (along t).
// ---------------------------------------------------------------------------
__global__ __launch_bounds__(256)
void aft_kve_transpose(const float* __restrict__ proj, const float* __restrict__ Kmax,
                       unsigned short* __restrict__ KVEt, int T, int D)
{
  __shared__ unsigned short tKV[32][33];
  __shared__ unsigned short tKE[32][33];
  const int tx = threadIdx.x, ty = threadIdx.y;
  const int t0 = blockIdx.x * 32, d0 = blockIdx.y * 32, b = blockIdx.z;
  #pragma unroll
  for (int i = 0; i < 4; ++i) {
    int t = t0 + ty + 8 * i;
    long long bt = (long long)b * T + t;
    const float* row = proj + bt * (3LL * D);
    float k = row[D + d0 + tx];
    float v = row[2 * D + d0 + tx];
    float ke = expf(k - Kmax[bt]);
    tKV[tx][ty + 8 * i] = f32_to_bf16_bits(ke * v);
    tKE[tx][ty + 8 * i] = f32_to_bf16_bits(ke);
  }
  __syncthreads();
  unsigned short* outb = KVEt + (long long)b * (2LL * D) * T;
  #pragma unroll
  for (int i = 0; i < 4; ++i) {
    int d = d0 + ty + 8 * i;
    outb[(long long)d * T + t0 + tx]       = tKV[ty + 8 * i][tx];
    outb[(long long)(D + d) * T + t0 + tx] = tKE[ty + 8 * i][tx];
  }
}

// ---------------------------------------------------------------------------
// EB[i][j] = (j<=i) ? exp(pos_bias[i][j] - rowmax_{j<=i}) : 0   (bf16, row-major A)
// ---------------------------------------------------------------------------
__global__ __launch_bounds__(256)
void aft_prep_bias(const float* __restrict__ pos_bias, unsigned short* __restrict__ EB,
                   int T, int ldpb)
{
  const int i = blockIdx.x, tid = threadIdx.x;
  const float* row = pos_bias + (long long)i * ldpb;

  float m = -INFINITY;
  for (int j = tid; j <= i; j += 256) m = fmaxf(m, row[j]);
  __shared__ float red[256];
  red[tid] = m; __syncthreads();
  for (int s = 128; s > 0; s >>= 1) {
    if (tid < s) red[tid] = fmaxf(red[tid], red[tid + s]);
    __syncthreads();
  }
  m = red[0];

  unsigned short* out = EB + (long long)i * T;
  for (int j = tid; j < T; j += 256)
    out[j] = (j <= i) ? f32_to_bf16_bits(expf(row[j] - m)) : (unsigned short)0;
}

// ---------------------------------------------------------------------------
// y = sigmoid(q) * num/denom  -> bf16 (NT loads: proj & NumDen are read-once)
// ---------------------------------------------------------------------------
__global__ __launch_bounds__(256)
void aft_combine(const float* __restrict__ proj, const float* __restrict__ NumDen,
                 unsigned short* __restrict__ Ybf, int D, long long total)
{
  long long idx = (long long)blockIdx.x * 256 + threadIdx.x;
  if (idx >= total) return;
  long long bt = idx / D;
  int d = (int)(idx - bt * D);
  float q   = __builtin_nontemporal_load(&proj[bt * (3LL * D) + d]);
  float num = __builtin_nontemporal_load(&NumDen[bt * (2LL * D) + d]);
  float den = __builtin_nontemporal_load(&NumDen[bt * (2LL * D) + D + d]);
  float sig = 1.0f / (1.0f + expf(-q));
  Ybf[idx] = f32_to_bf16_bits(sig * num / den);
}

// ---------------------------------------------------------------------------
extern "C" void kernel_launch(void* const* d_in, const int* in_sizes, int n_in,
                              void* d_out, int out_size, void* d_ws, size_t ws_size,
                              hipStream_t stream) {
  (void)in_sizes; (void)n_in; (void)out_size; (void)ws_size;
  const float* x        = (const float*)d_in[0];   // [8,2048,1024]
  const float* w_attn   = (const float*)d_in[1];   // [1024,3072]
  const float* w_proj   = (const float*)d_in[2];   // [1024,1024]
  const float* pos_bias = (const float*)d_in[3];   // [2048,2048]
  float* out = (float*)d_out;                      // [8,2048,1024]

  const int Bb = 8, T = 2048, D = 1024;
  const long long BT = (long long)Bb * T;          // 16384

  size_t off = 0;
  auto wsa = [&](size_t bytes) -> void* {
    off = (off + 255) & ~(size_t)255;
    void* p = (char*)d_ws + off;
    off += bytes;
    return p;
  };
  unsigned short* Xbf   = (unsigned short*)wsa(BT * D * 2);             // 32 MB
  unsigned short* WabT  = (unsigned short*)wsa((size_t)D * 3 * D * 2);  //  6 MB  [3D][D]
  unsigned short* WpbT  = (unsigned short*)wsa((size_t)D * D * 2);      //  2 MB  [D][D]
  float*          proj  = (float*)wsa(BT * 3 * D * 4);                  // 192 MB
  float*          Kmax  = (float*)wsa(BT * 4);                          // 64 KB
  unsigned short* KVEt  = (unsigned short*)wsa(BT * 2 * D * 2);         // 64 MB  [b][2D][T]
  unsigned short* EB    = (unsigned short*)wsa((size_t)T * T * 2);      //  8 MB
  float*         NumDen = (float*)wsa(BT * 2 * D * 4);                  // 128 MB
  unsigned short* Ybf   = (unsigned short*)wsa(BT * D * 2);             // 32 MB

  // 1) operand converts (weights transposed to [N][K] for K-contiguous B)
  aft_cvt_bf16<<<(unsigned)((BT * D + 255) / 256), 256, 0, stream>>>(x, Xbf, BT * D);
  aft_transpose_cvt<<<dim3(3 * D / 32, D / 32), dim3(32, 8), 0, stream>>>(w_attn, WabT, D, 3 * D);
  aft_transpose_cvt<<<dim3(D / 32, D / 32), dim3(32, 8), 0, stream>>>(w_proj, WpbT, D, D);

  // 2) proj = x @ w_attn : [16384,1024] x [1024,3072]
  aft_gemm_bf16_wmma<<<dim3(3 * D / BN, (unsigned)(BT / BM), 1), 256, 0, stream>>>(
      Xbf, WabT, proj, D, D, D, 3 * D, 0, 0, 0, 0);

  // 3) rowmax(k), fused exp + transpose into KVEt, triangular exp-bias
  aft_rowmax_k<<<dim3(T, Bb), 256, 0, stream>>>(proj, Kmax, T, D);
  aft_kve_transpose<<<dim3(T / 32, D / 32, Bb), dim3(32, 8), 0, stream>>>(proj, Kmax, KVEt, T, D);
  aft_prep_bias<<<T, 256, 0, stream>>>(pos_bias, EB, T, T);

  // 4) [num|denom] = EB @ KVE, batched over B, triangular K-clip
  aft_gemm_bf16_wmma<<<dim3(2 * D / BN, T / BM, Bb), 256, 0, stream>>>(
      EB, KVEt, NumDen, T, T, T, 2 * D,
      0, (long long)(2 * D) * T, (long long)T * 2 * D, 1);

  // 5) y = sigmoid(q) * num/denom (bf16)
  aft_combine<<<(unsigned)((BT * D + 255) / 256), 256, 0, stream>>>(proj, NumDen, Ybf, D, BT * D);

  // 6) out = y @ w_proj : [16384,1024] x [1024,1024] (f32 out)
  aft_gemm_bf16_wmma<<<dim3(D / BN, (unsigned)(BT / BM), 1), 256, 0, stream>>>(
      Ybf, WpbT, out, D, D, D, D, 0, 0, 0, 0);
}
